// Encoder_42202348650556
// MI455X (gfx1250) — compile-verified, matrix operand order checked by reading, not run on hardware
//
#include <hip/hip_runtime.h>
#include <hip/hip_bf16.h>

// ---------------------------------------------------------------------------
// CDNA5 (gfx1250) encoder: all big matmuls on v_wmma_f32_16x16x32_bf16.
// M-tile 64 fused 2-layer MLPs, async global->LDS staging when available.
// ---------------------------------------------------------------------------

typedef __attribute__((ext_vector_type(16))) __bf16 v16bf;
typedef __attribute__((ext_vector_type(8)))  float  f32x8;
typedef int v4i_native __attribute__((vector_size(16)));

#define AS1 __attribute__((address_space(1)))
#define AS3 __attribute__((address_space(3)))

union Frag {
    uint4 u[2];
    v16bf v;
};

__device__ inline f32x8 wmma_bf16(v16bf a, v16bf b, f32x8 c) {
    // 8 args: (neg_a, A, neg_b, B, c_mod, C, reuse_a, reuse_b)
    return __builtin_amdgcn_wmma_f32_16x16x32_bf16(false, a, false, b, (short)0, c, false, false);
}

__device__ inline float gelu_f(float x) {
    const float k0 = 0.7978845608028654f; // sqrt(2/pi)
    float x3 = x * x * x;
    return 0.5f * x * (1.0f + tanhf(k0 * (x + 0.044715f * x3)));
}

#if defined(__has_builtin)
#if __has_builtin(__builtin_amdgcn_global_load_async_to_lds_b128)
#define HAVE_ASYNC_LDS 1
#endif
#if __has_builtin(__builtin_amdgcn_s_wait_asynccnt)
#define WAIT_ASYNC() __builtin_amdgcn_s_wait_asynccnt(0)
#endif
#endif
#ifndef WAIT_ASYNC
#define WAIT_ASYNC() asm volatile("s_wait_asynccnt 0x0" ::: "memory")
#endif

// ---------------------------------------------------------------------------
// x passthrough copy (float4)
// ---------------------------------------------------------------------------
__global__ __launch_bounds__(256) void copy_f32x4(const float4* __restrict__ src,
                                                  float4* __restrict__ dst, int n4) {
    int i = blockIdx.x * 256 + threadIdx.x;
    if (i < n4) dst[i] = src[i];
}

// ---------------------------------------------------------------------------
// Weight prep: f32 [K][N] (z slices) -> bf16 [N][K]
// ---------------------------------------------------------------------------
__global__ __launch_bounds__(256) void transpose_bf16(const float* __restrict__ src,
                                                      __hip_bfloat16* __restrict__ dst,
                                                      int K, int N) {
    int sl = blockIdx.z;
    src += (size_t)sl * K * N;
    dst += (size_t)sl * K * N;
    int idx = blockIdx.x * 256 + threadIdx.x;
    if (idx < K * N) {
        int n = idx / K, k = idx % K;
        dst[idx] = __float2bfloat16(src[(size_t)k * N + n]);
    }
}

// ---------------------------------------------------------------------------
// conv0: (16,256,256,3) -> gelu(conv3x3 s2 + b) as bf16 (16,128,128,64)
// ---------------------------------------------------------------------------
__global__ __launch_bounds__(256) void conv0_kernel(const float* __restrict__ x,
                                                    const float* __restrict__ w,
                                                    const float* __restrict__ b,
                                                    __hip_bfloat16* __restrict__ g0) {
    size_t idx = (size_t)blockIdx.x * 256 + threadIdx.x;
    if (idx >= (size_t)16 * 128 * 128 * 64) return;
    int co = (int)(idx & 63);
    size_t p  = idx >> 6;
    int xo = (int)(p & 127);
    size_t p2 = p >> 7;
    int yo = (int)(p2 & 127);
    int bi = (int)(p2 >> 7);
    float acc = b[co];
    #pragma unroll
    for (int dy = 0; dy < 3; ++dy) {
        int yi = 2 * yo + dy;
        if (yi >= 256) continue;
        #pragma unroll
        for (int dx = 0; dx < 3; ++dx) {
            int xi = 2 * xo + dx;
            if (xi >= 256) continue;
            const float* px = &x[(((size_t)bi * 256 + yi) * 256 + xi) * 3];
            const float* pw = &w[(dy * 3 + dx) * 3 * 64 + co];
            acc += px[0] * pw[0] + px[1] * pw[64] + px[2] * pw[128];
        }
    }
    g0[idx] = __float2bfloat16(gelu_f(acc));
}

// ---------------------------------------------------------------------------
// Fused 2-layer MLP:  out = (act(A @ W1 + b1)) @ W2 + b2
//   A   : [M][K1] bf16        W1t : [N1][K1] bf16 (transposed)
//   W2t : [N2][N1] bf16       out f32 and/or bf16 (nullable)
// EPI: 0 = relu, 1 = gelu.  Block: 256 threads (8 waves), M tile = 64.
// ---------------------------------------------------------------------------
template <int K1, int N1, int N2, int EPI>
__global__ __launch_bounds__(256) void fused_mlp(const __hip_bfloat16* __restrict__ A,
                                                 const __hip_bfloat16* __restrict__ W1t,
                                                 const float* __restrict__ b1,
                                                 const __hip_bfloat16* __restrict__ W2t,
                                                 const float* __restrict__ b2,
                                                 float* __restrict__ outF,
                                                 __hip_bfloat16* __restrict__ outB) {
    __shared__ alignas(16) __hip_bfloat16 sA[64 * K1];
    __shared__ alignas(16) __hip_bfloat16 sH[64 * N1];

    const int tid  = threadIdx.x;
    const int wave = tid >> 5;
    const int lane = tid & 31;
    const int m0   = blockIdx.x * 64;

    // ---- stage A tile (64 contiguous rows) into LDS --------------------------
#if defined(HAVE_ASYNC_LDS)
    {
        const char* g = (const char*)(A + (size_t)m0 * K1);
        char* s = (char*)sA;
        for (int i = tid * 16; i < 64 * K1 * 2; i += 256 * 16) {
            __builtin_amdgcn_global_load_async_to_lds_b128(
                (AS1 v4i_native*)(uintptr_t)(g + i),
                (AS3 v4i_native*)(AS3 void*)(void*)(s + i),
                0, 0);
        }
        WAIT_ASYNC();
    }
#else
    {
        const uint4* gA = (const uint4*)(A + (size_t)m0 * K1);
        uint4* sA4 = (uint4*)sA;
        for (int i = tid; i < 64 * K1 / 8; i += 256) sA4[i] = gA[i];
    }
#endif
    __syncthreads();

    const int nl = lane & 15;          // M (A) / N (B,C) lane index
    const int kb = (lane >> 4) << 3;   // K half select for A/B fragments
    const int mh = (lane >> 4) << 3;   // C/D row-half select

    // ---------------- phase 1: H = act(A @ W1 + b1) -> LDS bf16 --------------
    for (int nt = wave; nt < N1 / 16; nt += 8) {
        const int n = nt * 16 + nl;
        if (nt + 8 < N1 / 16)
            __builtin_prefetch(&W1t[(size_t)(n + 128) * K1], 0, 0);
        f32x8 acc[4] = {};
        #pragma unroll
        for (int ks = 0; ks < K1 / 32; ++ks) {
            Frag bf;
            const __hip_bfloat16* pb = &W1t[(size_t)n * K1 + ks * 32 + kb];
            bf.u[0] = *(const uint4*)pb;
            bf.u[1] = *(const uint4*)(pb + 16);
            #pragma unroll
            for (int mt = 0; mt < 4; ++mt) {
                Frag af;
                const __hip_bfloat16* pa = &sA[(mt * 16 + nl) * K1 + ks * 32 + kb];
                af.u[0] = *(const uint4*)pa;
                af.u[1] = *(const uint4*)(pa + 16);
                acc[mt] = wmma_bf16(af.v, bf.v, acc[mt]);
            }
        }
        const float bias = b1[n];
        #pragma unroll
        for (int mt = 0; mt < 4; ++mt) {
            #pragma unroll
            for (int r = 0; r < 8; ++r) {
                float v = acc[mt][r] + bias;
                v = (EPI == 0) ? fmaxf(v, 0.0f) : gelu_f(v);
                sH[(mt * 16 + mh + r) * N1 + n] = __float2bfloat16(v);
            }
        }
    }
    __syncthreads();

    // ---------------- phase 2: out = H @ W2 + b2 -----------------------------
    const int NJ = 4 * (N2 / 16);
    for (int job = wave; job < NJ; job += 8) {
        const int mt = job & 3;
        const int nt = job >> 2;
        const int n  = nt * 16 + nl;
        f32x8 acc = {};
        #pragma unroll
        for (int ks = 0; ks < N1 / 32; ++ks) {
            Frag af, bf;
            const __hip_bfloat16* pa = &sH[(mt * 16 + nl) * N1 + ks * 32 + kb];
            af.u[0] = *(const uint4*)pa;
            af.u[1] = *(const uint4*)(pa + 16);
            const __hip_bfloat16* pb = &W2t[(size_t)n * N1 + ks * 32 + kb];
            bf.u[0] = *(const uint4*)pb;
            bf.u[1] = *(const uint4*)(pb + 16);
            acc = wmma_bf16(af.v, bf.v, acc);
        }
        const float bias = b2[n];
        if (outF) {
            #pragma unroll
            for (int r = 0; r < 8; ++r) {
                size_t gm = (size_t)(m0 + mt * 16 + mh + r);
                outF[gm * N2 + n] = acc[r] + bias;
            }
        }
        if (outB) {
            #pragma unroll
            for (int r = 0; r < 8; ++r) {
                size_t gm = (size_t)(m0 + mt * 16 + mh + r);
                outB[gm * N2 + n] = __float2bfloat16(acc[r] + bias);
            }
        }
    }
}

// ---------------------------------------------------------------------------
// conv1: implicit GEMM, a1 f32 (16,128,128,64) -> g1 = gelu(conv+b) bf16
// (16,64,64,128).  Block = (x-tile of 16 outputs, one y row, one batch).
// ---------------------------------------------------------------------------
__global__ __launch_bounds__(256) void conv1_kernel(const float* __restrict__ a1,
                                                    const __hip_bfloat16* __restrict__ wt, // [9][128][64]
                                                    const float* __restrict__ bias,
                                                    __hip_bfloat16* __restrict__ g1) {
    __shared__ alignas(16) __hip_bfloat16 sIn[3][34][64];
    const int xt = blockIdx.x;   // 0..3
    const int yo = blockIdx.y;   // 0..63
    const int bi = blockIdx.z;   // 0..15
    const int tid  = threadIdx.x;
    const int wave = tid >> 5;
    const int lane = tid & 31;
    const int x0 = xt * 16;

    for (int i = tid; i < 3 * 34 * 16; i += 256) {
        int dy  = i / (34 * 16);
        int rem = i % (34 * 16);
        int xi  = rem / 16;
        int cg  = rem % 16;
        int yi  = 2 * yo + dy;
        int xg  = 2 * x0 + xi;
        float4 v = make_float4(0.f, 0.f, 0.f, 0.f);
        if (yi < 128 && xg < 128)
            v = *(const float4*)&a1[(((size_t)bi * 128 + yi) * 128 + xg) * 64 + cg * 4];
        __hip_bfloat16* d = &sIn[dy][xi][cg * 4];
        d[0] = __float2bfloat16(v.x);
        d[1] = __float2bfloat16(v.y);
        d[2] = __float2bfloat16(v.z);
        d[3] = __float2bfloat16(v.w);
    }
    __syncthreads();

    const int nl = lane & 15;
    const int kb = (lane >> 4) << 3;
    const int mh = (lane >> 4) << 3;

    f32x8 acc = {};
    for (int dy = 0; dy < 3; ++dy) {
        for (int dx = 0; dx < 3; ++dx) {
            const __hip_bfloat16* wbase =
                wt + ((size_t)(dy * 3 + dx) * 128 + wave * 16 + nl) * 64;
            #pragma unroll
            for (int ks = 0; ks < 2; ++ks) {
                Frag af, bf;
                const __hip_bfloat16* pa = &sIn[dy][2 * nl + dx][ks * 32 + kb];
                af.u[0] = *(const uint4*)pa;
                af.u[1] = *(const uint4*)(pa + 16);
                const __hip_bfloat16* pb = wbase + ks * 32 + kb;
                bf.u[0] = *(const uint4*)pb;
                bf.u[1] = *(const uint4*)(pb + 16);
                acc = wmma_bf16(af.v, bf.v, acc);
            }
        }
    }
    const int co = wave * 16 + nl;
    const float bv = bias[co];
    #pragma unroll
    for (int r = 0; r < 8; ++r) {
        int xo = x0 + mh + r;
        float c = acc[r] + bv;
        g1[(((size_t)bi * 64 + yo) * 64 + xo) * 128 + co] = __float2bfloat16(gelu_f(c));
    }
}

// ---------------------------------------------------------------------------
// Build attention memory tensor: mem[b,nx,ny,m,ch] bf16
// ---------------------------------------------------------------------------
__global__ __launch_bounds__(256) void mem_build(const float* __restrict__ a2,
                                                 const float* __restrict__ pos,
                                                 __hip_bfloat16* __restrict__ mem) {
    size_t idx = (size_t)blockIdx.x * 256 + threadIdx.x;
    if (idx >= (size_t)16 * 8 * 8 * 64 * 160) return;
    int ch = (int)(idx % 160);
    size_t t = idx / 160;
    int m  = (int)(t & 63);
    size_t t2 = t >> 6;
    int ny = (int)(t2 & 7);
    size_t t3 = t2 >> 3;
    int nx = (int)(t3 & 7);
    int bi = (int)(t3 >> 3);
    int w1 = m >> 3, w2 = m & 7;
    float v;
    if (ch < 128)
        v = a2[(((size_t)bi * 64 + (w1 * 8 + nx)) * 64 + (w2 * 8 + ny)) * 128 + ch];
    else
        v = pos[((size_t)w1 * 8 + w2) * 32 + (ch - 128)];
    mem[idx] = __float2bfloat16(v);
}

__global__ __launch_bounds__(256) void qa_gather(const float* __restrict__ a2,
                                                 __hip_bfloat16* __restrict__ qa) {
    int idx = blockIdx.x * 256 + threadIdx.x;
    if (idx >= 1024 * 128) return;
    int c = idx & 127;
    int w = idx >> 7;
    int ny = w & 7, nx = (w >> 3) & 7, bi = w >> 6;
    qa[idx] = __float2bfloat16(
        a2[(((size_t)bi * 64 + (32 + nx)) * 64 + (32 + ny)) * 128 + c]);
}

// ---------------------------------------------------------------------------
// Attention combine: per window (1024): softmax(q.k/8) @ v -> wv bf16
// ---------------------------------------------------------------------------
__global__ __launch_bounds__(64) void attn_combine(const __hip_bfloat16* __restrict__ qb, // [1024][32]
                                                   const __hip_bfloat16* __restrict__ kb, // [65536][32]
                                                   const __hip_bfloat16* __restrict__ vb, // [65536][128]
                                                   __hip_bfloat16* __restrict__ wvb) {   // [1024][128]
    const int w = blockIdx.x;
    const int t = threadIdx.x; // 0..63
    __shared__ float sQ[32];
    __shared__ float sP[64];
    __shared__ float red[64];

    if (t < 32) sQ[t] = __bfloat162float(qb[(size_t)w * 32 + t]);
    __syncthreads();

    const __hip_bfloat16* kr = &kb[((size_t)w * 64 + t) * 32];
    float s = 0.0f;
    #pragma unroll
    for (int c = 0; c < 32; ++c) s += sQ[c] * __bfloat162float(kr[c]);
    s *= 0.125f;

    red[t] = s;
    __syncthreads();
    float mx = red[0];
    for (int i = 1; i < 64; ++i) mx = fmaxf(mx, red[i]);
    __syncthreads();
    float e = __expf(s - mx);
    red[t] = e;
    __syncthreads();
    float sum = 0.0f;
    for (int i = 0; i < 64; ++i) sum += red[i];
    sP[t] = e / sum;
    __syncthreads();

    #pragma unroll
    for (int cc = 0; cc < 2; ++cc) {
        int c = t * 2 + cc;
        float acc = 0.0f;
        for (int m = 0; m < 64; ++m)
            acc += sP[m] * __bfloat162float(vb[((size_t)w * 64 + m) * 128 + c]);
        wvb[(size_t)w * 128 + c] = __float2bfloat16(acc);
    }
}

// ---------------------------------------------------------------------------
// Host side
// ---------------------------------------------------------------------------
extern "C" void kernel_launch(void* const* d_in, const int* in_sizes, int n_in,
                              void* d_out, int out_size, void* d_ws, size_t ws_size,
                              hipStream_t stream) {
    (void)in_sizes; (void)n_in; (void)out_size; (void)ws_size;

    const float* x       = (const float*)d_in[0];
    const float* conv0_w = (const float*)d_in[1];
    const float* conv0_b = (const float*)d_in[2];
    const float* d0a_w   = (const float*)d_in[3];
    const float* d0a_b   = (const float*)d_in[4];
    const float* d0b_w   = (const float*)d_in[5];
    const float* d0b_b   = (const float*)d_in[6];
    const float* conv1_w = (const float*)d_in[7];
    const float* conv1_b = (const float*)d_in[8];
    const float* d1a_w   = (const float*)d_in[9];
    const float* d1a_b   = (const float*)d_in[10];
    const float* d1b_w   = (const float*)d_in[11];
    const float* d1b_b   = (const float*)d_in[12];
    const float* pos_enc = (const float*)d_in[13];
    const float* q1_w = (const float*)d_in[14];
    const float* q1_b = (const float*)d_in[15];
    const float* q2_w = (const float*)d_in[16];
    const float* q2_b = (const float*)d_in[17];
    const float* k1_w = (const float*)d_in[18];
    const float* k1_b = (const float*)d_in[19];
    const float* k2_w = (const float*)d_in[20];
    const float* k2_b = (const float*)d_in[21];
    const float* v1_w = (const float*)d_in[22];
    const float* v1_b = (const float*)d_in[23];
    const float* v2_w = (const float*)d_in[24];
    const float* v2_b = (const float*)d_in[25];
    const float* o1_w = (const float*)d_in[26];
    const float* o1_b = (const float*)d_in[27];
    const float* o2_w = (const float*)d_in[28];
    const float* o2_b = (const float*)d_in[29];
    const float* m1_w = (const float*)d_in[30];
    const float* m1_b = (const float*)d_in[31];
    const float* m2_w = (const float*)d_in[32];
    const float* m2_b = (const float*)d_in[33];

    float* out = (float*)d_out;
    const size_t o_x  = 0;
    const size_t o_a1 = 3145728;             // 16*256*256*3
    const size_t o_a2 = o_a1 + 16777216;     // 16*128*128*64
    const size_t o_a3 = o_a2 + 8388608;      // 16*64*64*128
    const size_t o_a4 = o_a3 + 131072;       // 16*8*8*128

    char* ws = (char*)d_ws;
    size_t off = 0;
    auto alloc = [&](size_t bytes) -> char* {
        char* p = ws + off;
        off = (off + bytes + 255) & ~(size_t)255;
        return p;
    };
    typedef __hip_bfloat16 bf;
    bf* g0   = (bf*)alloc((size_t)16 * 128 * 128 * 64 * 2);
    bf* g1   = (bf*)alloc((size_t)16 * 64 * 64 * 128 * 2);
    bf* memb = (bf*)alloc((size_t)16 * 8 * 8 * 64 * 160 * 2);
    bf* qa   = (bf*)alloc((size_t)1024 * 128 * 2);
    bf* kout = (bf*)alloc((size_t)65536 * 32 * 2);
    bf* vout = (bf*)alloc((size_t)65536 * 128 * 2);
    bf* qout = (bf*)alloc((size_t)1024 * 32 * 2);
    bf* wv   = (bf*)alloc((size_t)1024 * 128 * 2);
    bf* a3b  = (bf*)alloc((size_t)1024 * 128 * 2);
    bf* d0at = (bf*)alloc((size_t)64 * 256 * 2);
    bf* d0bt = (bf*)alloc((size_t)256 * 64 * 2);
    bf* d1at = (bf*)alloc((size_t)128 * 512 * 2);
    bf* d1bt = (bf*)alloc((size_t)512 * 128 * 2);
    bf* c1wt = (bf*)alloc((size_t)9 * 128 * 64 * 2);
    bf* q1t  = (bf*)alloc((size_t)128 * 256 * 2);
    bf* q2t  = (bf*)alloc((size_t)256 * 32 * 2);
    bf* k1t  = (bf*)alloc((size_t)160 * 256 * 2);
    bf* k2t  = (bf*)alloc((size_t)256 * 32 * 2);
    bf* v1t  = (bf*)alloc((size_t)160 * 256 * 2);
    bf* v2t  = (bf*)alloc((size_t)256 * 128 * 2);
    bf* o1t  = (bf*)alloc((size_t)128 * 256 * 2);
    bf* o2t  = (bf*)alloc((size_t)256 * 128 * 2);
    bf* m1t  = (bf*)alloc((size_t)128 * 256 * 2);
    bf* m2t  = (bf*)alloc((size_t)256 * 64 * 2);

    auto tr = [&](const float* src, bf* dst, int K, int N, int slices) {
        dim3 g((K * N + 255) / 256, 1, slices);
        transpose_bf16<<<g, 256, 0, stream>>>(src, dst, K, N);
    };

    // ---- weight prep ----
    tr(d0a_w, d0at, 64, 256, 1);
    tr(d0b_w, d0bt, 256, 64, 1);
    tr(d1a_w, d1at, 128, 512, 1);
    tr(d1b_w, d1bt, 512, 128, 1);
    tr(conv1_w, c1wt, 64, 128, 9);
    tr(q1_w, q1t, 128, 256, 1);
    tr(q2_w, q2t, 256, 32, 1);
    tr(k1_w, k1t, 160, 256, 1);
    tr(k2_w, k2t, 256, 32, 1);
    tr(v1_w, v1t, 160, 256, 1);
    tr(v2_w, v2t, 256, 128, 1);
    tr(o1_w, o1t, 128, 256, 1);
    tr(o2_w, o2t, 256, 128, 1);
    tr(m1_w, m1t, 128, 256, 1);
    tr(m2_w, m2t, 256, 64, 1);

    // ---- output 0: x passthrough ----
    copy_f32x4<<<(786432 + 255) / 256, 256, 0, stream>>>(
        (const float4*)x, (float4*)(out + o_x), 786432);

    // ---- block 1 ----
    conv0_kernel<<<(16 * 128 * 128 * 64) / 256, 256, 0, stream>>>(x, conv0_w, conv0_b, g0);
    fused_mlp<64, 256, 64, 0><<<262144 / 64, 256, 0, stream>>>(
        g0, d0at, d0a_b, d0bt, d0b_b, out + o_a1, nullptr);

    // ---- block 2 ----
    {
        dim3 g(4, 64, 16);
        conv1_kernel<<<g, 256, 0, stream>>>(out + o_a1, c1wt, conv1_b, g1);
    }
    fused_mlp<128, 512, 128, 0><<<65536 / 64, 256, 0, stream>>>(
        g1, d1at, d1a_b, d1bt, d1b_b, out + o_a2, nullptr);

    // ---- attention ----
    mem_build<<<(16 * 8 * 8 * 64 * 160) / 256, 256, 0, stream>>>(out + o_a2, pos_enc, memb);
    qa_gather<<<(1024 * 128) / 256, 256, 0, stream>>>(out + o_a2, qa);

    fused_mlp<160, 256, 32, 1><<<65536 / 64, 256, 0, stream>>>(
        memb, k1t, k1_b, k2t, k2_b, nullptr, kout);
    fused_mlp<160, 256, 128, 1><<<65536 / 64, 256, 0, stream>>>(
        memb, v1t, v1_b, v2t, v2_b, nullptr, vout);
    fused_mlp<128, 256, 32, 1><<<1024 / 64, 256, 0, stream>>>(
        qa, q1t, q1_b, q2t, q2_b, nullptr, qout);

    attn_combine<<<1024, 64, 0, stream>>>(qout, kout, vout, wv);

    fused_mlp<128, 256, 128, 1><<<1024 / 64, 256, 0, stream>>>(
        wv, o1t, o1_b, o2t, o2_b, out + o_a3, a3b);

    // ---- final MLP ----
    fused_mlp<128, 256, 64, 1><<<1024 / 64, 256, 0, stream>>>(
        a3b, m1t, m1_b, m2t, m2_b, out + o_a4, nullptr);
}
